// ConvSNN1_31198642438227
// MI455X (gfx1250) — matrix operand order repaired and checked by reading, not compiled, not run on hardware
//
#include <hip/hip_runtime.h>
#include <hip/hip_bf16.h>

typedef __attribute__((ext_vector_type(16))) _Float16 v16h;
typedef __attribute__((ext_vector_type(8)))  _Float16 v8h;
typedef __attribute__((ext_vector_type(8)))  float    v8f;

// ---- problem constants ----
#define T_STEPS 32
#define BATCH   128
#define C1      20
#define H1      26          // conv1 output spatial
#define HW1     (H1*H1)     // 676
#define PDIM    13          // pooled spatial
#define PHW     (PDIM*PDIM) // 169
#define C2      50
#define H2      11
#define HW2     (H2*H2)     // 121
#define K1      6050        // 50*121
#define KP      6080        // K padded to multiple of 32
#define HID     500
#define NP      512         // HID padded to multiple of 16
#define NOUT    10

#define DTM 0.1f            // dt * tau_mem_inv
#define DTS 0.2f            // dt * tau_syn_inv

// async 16B copy: global -> LDS via CDNA5 async path (ASYNCcnt tracked)
__device__ __forceinline__ void async_copy_b128(void* lds_dst, const void* gsrc) {
    unsigned lds_off = (unsigned)(size_t)lds_dst;   // LDS-space offset
    asm volatile("global_load_async_to_lds_b128 %0, %1, off"
                 :: "v"(lds_off), "v"(gsrc) : "memory");
}
__device__ __forceinline__ void async_wait0() {
    asm volatile("s_wait_asynccnt 0" ::: "memory");
}

// ------------------------------------------------------------------
// zero workspace (states + spike pads must start at 0 every call)
// ------------------------------------------------------------------
__global__ void zero_ws_kernel(unsigned int* __restrict__ ws, size_t n_words) {
    size_t i = (size_t)blockIdx.x * blockDim.x + threadIdx.x;
    size_t stride = (size_t)gridDim.x * blockDim.x;
    for (; i < n_words; i += stride) ws[i] = 0u;
}

// ------------------------------------------------------------------
// pack w_fc1 [500][6050] f32 -> [512][6080] f16, zero padded
// ------------------------------------------------------------------
__global__ void pack_wfc1_kernel(const float* __restrict__ w,
                                 _Float16* __restrict__ wh) {
    int idx = blockIdx.x * 256 + threadIdx.x;
    int k = idx % KP;
    int n = idx / KP;
    float v = (n < HID && k < K1) ? w[n * K1 + k] : 0.0f;
    wh[idx] = (_Float16)v;
}

// ------------------------------------------------------------------
// Fused conv stage: one block per batch image (128 blocks x 1024).
//   stage 0: x_t[b], w_conv1, w_conv2 -> LDS via async-to-LDS b128
//   stage 1: conv1(1->20,3x3) + LIF0 (global state) + 2x2 avgpool -> LDS p
//   stage 2: conv2(20->50,3x3) from LDS + LIF1 (global state)
//            -> f16 spikes into padded A-matrix buffer s1 [B][6080]
// LDS: 784 + 180 + 9000 + 3380 floats = ~53.4 KB (of 320 KB/WGP)
// ------------------------------------------------------------------
__global__ void __launch_bounds__(1024)
conv_fused_kernel(const float* __restrict__ x_t,     // [B][1][28][28]
                  const float* __restrict__ w1g,     // [20][1][3][3]
                  const float* __restrict__ w2g,     // [50][20][3][3]
                  float* __restrict__ v0,            // [B][20][26][26]
                  float* __restrict__ i0,
                  float* __restrict__ v1,            // [B][6050]
                  float* __restrict__ i1,
                  _Float16* __restrict__ s1)         // [B][6080]
{
    __shared__ __align__(16) float x_lds[784];
    __shared__ __align__(16) float w1_lds[C1 * 9];        // 180
    __shared__ __align__(16) float w2_lds[C2 * C1 * 9];   // 9000
    __shared__ __align__(16) float p_lds[C1 * PHW];       // 3380

    const int tid = threadIdx.x;
    const int b   = blockIdx.x;

    // ---- stage 0: async bulk copy of inputs into LDS (16B per lane) ----
    const float* xb = x_t + b * 784;
    for (int i = tid; i < 784 / 4; i += 1024)
        async_copy_b128(&x_lds[4 * i], xb + 4 * i);
    for (int i = tid; i < (C1 * 9) / 4; i += 1024)
        async_copy_b128(&w1_lds[4 * i], w1g + 4 * i);
    for (int i = tid; i < (C2 * C1 * 9) / 4; i += 1024)
        async_copy_b128(&w2_lds[4 * i], w2g + 4 * i);
    async_wait0();
    __syncthreads();

    // ---- stage 1: conv1 + lif0 + pool -> p_lds ----
    for (int i = tid; i < C1 * PHW; i += 1024) {
        int px = i % PDIM;
        int py = (i / PDIM) % PDIM;
        int c  = i / PHW;
        const float* wk = &w1_lds[c * 9];
        float zsum = 0.0f;
#pragma unroll
        for (int dy = 0; dy < 2; ++dy) {
#pragma unroll
            for (int dx = 0; dx < 2; ++dx) {
                int y  = 2 * py + dy;
                int xq = 2 * px + dx;
                float a = 0.0f;
#pragma unroll
                for (int ky = 0; ky < 3; ++ky)
#pragma unroll
                    for (int kx = 0; kx < 3; ++kx)
                        a = fmaf(wk[ky * 3 + kx], x_lds[(y + ky) * 28 + (xq + kx)], a);
                int sidx = ((b * C1 + c) * H1 + y) * H1 + xq;
                float v   = v0[sidx];
                float cur = i0[sidx];
                float vdec = v + DTM * (cur - v);
                float idec = (1.0f - DTS) * cur;
                float z = (vdec - 0.1f) > 0.0f ? 1.0f : 0.0f;
                v0[sidx] = (1.0f - z) * vdec;
                i0[sidx] = idec + a;
                zsum += z;
            }
        }
        p_lds[i] = 0.25f * zsum;
    }
    __syncthreads();

    // ---- stage 2: conv2 + lif1 -> s1 (f16 spikes) ----
    for (int i = tid; i < K1; i += 1024) {
        int xq = i % H2;
        int y  = (i / H2) % H2;
        int oc = i / HW2;
        const float* wc = &w2_lds[oc * C1 * 9];
        float a = 0.0f;
        for (int ic = 0; ic < C1; ++ic) {
            const float* pi = &p_lds[ic * PHW];
            const float* wi = wc + ic * 9;
#pragma unroll
            for (int ky = 0; ky < 3; ++ky)
#pragma unroll
                for (int kx = 0; kx < 3; ++kx)
                    a = fmaf(wi[ky * 3 + kx], pi[(y + ky) * PDIM + (xq + kx)], a);
        }
        int sidx = b * K1 + i;
        float v   = v1[sidx];
        float cur = i1[sidx];
        float vdec = v + DTM * (cur - v);
        float idec = (1.0f - DTS) * cur;
        float z = (vdec - 0.05f) > 0.0f ? 1.0f : 0.0f;
        v1[sidx] = (1.0f - z) * vdec;
        i1[sidx] = idec + a;
        s1[b * KP + i] = (_Float16)z;
    }
}

// ------------------------------------------------------------------
// Fused head: fc1 WMMA GEMM + LIF2 + fc2 + LI readout.
// One block per 16-batch M-tile: 8 blocks x 256 threads (8 waves).
// Each wave computes a 16x64 C tile: 1 A fragment reused over 4 B
// fragments / 4 accumulators. Fragment loads are manually
// double-buffered so next-iteration loads overlap the 4-WMMA group.
// ------------------------------------------------------------------
__global__ void __launch_bounds__(256)
fc_head_kernel(const _Float16* __restrict__ A,     // [128][6080] spikes
               const _Float16* __restrict__ Bw,    // [512][6080] f16 weights
               float* __restrict__ v2,             // [128][500]
               float* __restrict__ i2,
               const float* __restrict__ w_fc2,    // [10][500]
               float* __restrict__ vo,             // [128][10]
               float* __restrict__ io,
               float* __restrict__ out_t)          // [128][10] slice of d_out
{
    __shared__ _Float16 s2_lds[16 * NP];           // 16 KB

    const int wave = threadIdx.x >> 5;             // 0..7 -> N group of 64
    const int lane = threadIdx.x & 31;
    const int tm   = blockIdx.x;                   // 0..7 -> 16-batch tile
    const int l16  = lane & 15;
    const int half = lane >> 4;

    const int rowA = tm * 16 + l16;
    const _Float16* ap = A + (size_t)rowA * KP + half * 16;

    const int n0 = wave * 64 + l16;
    const _Float16* bp0 = Bw + (size_t)(n0 +  0) * KP + half * 16;
    const _Float16* bp1 = Bw + (size_t)(n0 + 16) * KP + half * 16;
    const _Float16* bp2 = Bw + (size_t)(n0 + 32) * KP + half * 16;
    const _Float16* bp3 = Bw + (size_t)(n0 + 48) * KP + half * 16;

    v8f acc0 = {}, acc1 = {}, acc2 = {}, acc3 = {};

    // prologue: load k=0 fragments
    v16h af, b0, b1, b2, b3;
    {
        v8h alo = *(const v8h*)(ap);
        v8h ahi = *(const v8h*)(ap + 8);
#pragma unroll
        for (int j = 0; j < 8; ++j) { af[j] = alo[j]; af[8 + j] = ahi[j]; }
        b0 = *(const v16h*)(bp0);
        b1 = *(const v16h*)(bp1);
        b2 = *(const v16h*)(bp2);
        b3 = *(const v16h*)(bp3);
    }

    for (int k = 32; k < KP; k += 32) {
        // issue next-iteration loads (independent of current WMMAs)
        v8h alo = *(const v8h*)(ap + k);
        v8h ahi = *(const v8h*)(ap + k + 8);
        v16h nb0 = *(const v16h*)(bp0 + k);
        v16h nb1 = *(const v16h*)(bp1 + k);
        v16h nb2 = *(const v16h*)(bp2 + k);
        v16h nb3 = *(const v16h*)(bp3 + k);
        __builtin_prefetch(bp0 + k + 256, 0, 1);
        __builtin_prefetch(bp2 + k + 256, 0, 1);

        acc0 = __builtin_amdgcn_wmma_f32_16x16x32_f16(false, af, false, b0,
                                                      (short)0, acc0, false, false);
        acc1 = __builtin_amdgcn_wmma_f32_16x16x32_f16(false, af, false, b1,
                                                      (short)0, acc1, false, false);
        acc2 = __builtin_amdgcn_wmma_f32_16x16x32_f16(false, af, false, b2,
                                                      (short)0, acc2, false, false);
        acc3 = __builtin_amdgcn_wmma_f32_16x16x32_f16(false, af, false, b3,
                                                      (short)0, acc3, false, false);

        // rotate buffers
#pragma unroll
        for (int j = 0; j < 8; ++j) { af[j] = alo[j]; af[8 + j] = ahi[j]; }
        b0 = nb0; b1 = nb1; b2 = nb2; b3 = nb3;
    }
    // epilogue WMMAs for the last K block
    acc0 = __builtin_amdgcn_wmma_f32_16x16x32_f16(false, af, false, b0,
                                                  (short)0, acc0, false, false);
    acc1 = __builtin_amdgcn_wmma_f32_16x16x32_f16(false, af, false, b1,
                                                  (short)0, acc1, false, false);
    acc2 = __builtin_amdgcn_wmma_f32_16x16x32_f16(false, af, false, b2,
                                                  (short)0, acc2, false, false);
    acc3 = __builtin_amdgcn_wmma_f32_16x16x32_f16(false, af, false, b3,
                                                  (short)0, acc3, false, false);

    // fused LIF2 epilogue -> spikes into LDS
    v8f accs[4] = {acc0, acc1, acc2, acc3};
#pragma unroll
    for (int q = 0; q < 4; ++q) {
        int n = wave * 64 + q * 16 + l16;
#pragma unroll
        for (int r = 0; r < 8; ++r) {
            int bl = half * 8 + r;                 // local batch row 0..15
            float z = 0.0f;
            if (n < HID) {
                int idx = (tm * 16 + bl) * HID + n;
                float v   = v2[idx];
                float cur = i2[idx];
                float vdec = v + DTM * (cur - v);
                float idec = (1.0f - DTS) * cur;
                z = (vdec - 0.05f) > 0.0f ? 1.0f : 0.0f;
                v2[idx] = (1.0f - z) * vdec;
                i2[idx] = idec + accs[q][r];
            }
            s2_lds[bl * NP + n] = (_Float16)z;
        }
    }
    __syncthreads();

    // fc2 + LI readout from LDS (160 of 256 threads)
    if (threadIdx.x < 16 * NOUT) {
        int o  = threadIdx.x % NOUT;
        int bl = threadIdx.x / NOUT;
        const _Float16* sb = &s2_lds[bl * NP];
        const float* wo = w_fc2 + o * HID;
        float a = 0.0f;
        for (int h = 0; h < HID; ++h)
            a = fmaf((float)sb[h], wo[h], a);
        int idx = (tm * 16 + bl) * NOUT + o;
        float v   = vo[idx];
        float cur = io[idx];
        float vnew = v + DTM * (cur - v);          // v_new uses pre-update i
        float inew = (1.0f - DTS) * cur + a;
        vo[idx] = vnew;
        io[idx] = inew;
        out_t[idx] = vnew;
    }
}

// ------------------------------------------------------------------
// host-side launch
// ------------------------------------------------------------------
static inline size_t align256(size_t x) { return (x + 255) & ~(size_t)255; }

extern "C" void kernel_launch(void* const* d_in, const int* in_sizes, int n_in,
                              void* d_out, int out_size, void* d_ws, size_t ws_size,
                              hipStream_t stream) {
    const float* x       = (const float*)d_in[0];  // [32][128][1][28][28]
    const float* w_conv1 = (const float*)d_in[1];  // [20][1][3][3]
    const float* w_conv2 = (const float*)d_in[2];  // [50][20][3][3]
    const float* w_fc1   = (const float*)d_in[3];  // [500][6050]
    const float* w_fc2   = (const float*)d_in[4];  // [10][500]
    float* out = (float*)d_out;                    // [32][128][10]

    // workspace layout (bytes, 256-aligned chunks)
    char* ws = (char*)d_ws;
    size_t off = 0;
    auto carve = [&](size_t bytes) -> char* {
        char* ptr = ws + off;
        off = align256(off + bytes);
        return ptr;
    };
    float* v0 = (float*)carve((size_t)BATCH * C1 * HW1 * 4);
    float* i0 = (float*)carve((size_t)BATCH * C1 * HW1 * 4);
    float* v1 = (float*)carve((size_t)BATCH * K1 * 4);
    float* i1 = (float*)carve((size_t)BATCH * K1 * 4);
    float* v2 = (float*)carve((size_t)BATCH * HID * 4);
    float* i2 = (float*)carve((size_t)BATCH * HID * 4);
    float* vo = (float*)carve((size_t)BATCH * NOUT * 4);
    float* io = (float*)carve((size_t)BATCH * NOUT * 4);
    _Float16* s1     = (_Float16*)carve((size_t)BATCH * KP * 2);
    _Float16* w_fc1h = (_Float16*)carve((size_t)NP * KP * 2);

    // 1) zero all state + spike pads (self-contained per call)
    size_t zero_words = off / 4;
    zero_ws_kernel<<<1024, 256, 0, stream>>>((unsigned int*)d_ws, zero_words);

    // 2) pack fc1 weights to padded f16 [512][6080]
    pack_wfc1_kernel<<<(NP * KP) / 256, 256, 0, stream>>>(w_fc1, w_fc1h);

    // 3) sequential time loop: 2 fused kernels per step
    for (int t = 0; t < T_STEPS; ++t) {
        const float* x_t = x + (size_t)t * BATCH * 784;
        conv_fused_kernel<<<BATCH, 1024, 0, stream>>>(
            x_t, w_conv1, w_conv2, v0, i0, v1, i1, s1);
        fc_head_kernel<<<8, 256, 0, stream>>>(
            s1, w_fc1h, v2, i2, w_fc2, vo, io, out + (size_t)t * BATCH * NOUT);
    }
}
